// MambaBranch_83090437308825
// MI455X (gfx1250) — matrix operand order repaired and checked by reading, not compiled, not run on hardware
//
#include <hip/hip_runtime.h>
#include <hip/hip_bf16.h>
#include <math.h>

// ---------------------------------------------------------------------------
// Mamba classifier, L == 1 specialization for MI455X (gfx1250, wave32, WMMA).
// GEMMs: v_wmma_f32_16x16x32_bf16, one 16x(16*NT) register tile per wave —
// A fragment loaded once per K-step and reused across NT WMMAs.
// ---------------------------------------------------------------------------

typedef __attribute__((ext_vector_type(16))) __bf16 v16bf;
typedef __attribute__((ext_vector_type(8)))  float  v8f;
typedef __attribute__((ext_vector_type(4)))  float  v4f;

// C[M,N] = A[M,K] (row major f32) x W[N,K]^T (row major f32) (+bias) (+Cin)
// grid = (N/(16*NT), M/128), block = 256 (8 waves); each wave: 16 x 16*NT tile.
template <int NT>
__global__ __launch_bounds__(256) void gemm_bf16_wmma(
    const float* __restrict__ A, const float* __restrict__ W,
    const float* __restrict__ bias, const float* __restrict__ Cin,
    float* __restrict__ Cout, int N, int K)
{
    const int lane  = threadIdx.x & 31;
    const int wave  = threadIdx.x >> 5;
    const int mTile = blockIdx.y * 8 + wave;
    const int nBase = blockIdx.x * (16 * NT);
    const int l15   = lane & 15;
    const bool hi   = lane >= 16;

    const float* arow = A + (size_t)((mTile << 4) + l15) * K;
    const float* wrow[NT];
    #pragma unroll
    for (int t = 0; t < NT; ++t)
        wrow[t] = W + (size_t)(nBase + t * 16 + l15) * K;

    // C/D layout: VGPR r holds row M = r + (hi ? 8 : 0), col N = lane&15
    const size_t crow = (size_t)((mTile << 4) + (hi ? 8 : 0)) * N;

    v8f acc[NT];
    if (Cin) {
        #pragma unroll
        for (int t = 0; t < NT; ++t) {
            const float* cb = Cin + crow + nBase + t * 16 + l15;
            #pragma unroll
            for (int r = 0; r < 8; ++r) acc[t][r] = cb[(size_t)r * N];
        }
    } else {
        #pragma unroll
        for (int t = 0; t < NT; ++t)
            #pragma unroll
            for (int r = 0; r < 8; ++r) acc[t][r] = 0.0f;
    }

    for (int k0 = 0; k0 < K; k0 += 32) {
        // A 16x32 bf16 layout: lane<16 -> K = {0..7, 16..23}; lane>=16 -> +8
        const int aoff = k0 + (hi ? 8 : 0);
        v4f a0 = *(const v4f*)(arow + aoff);
        v4f a1 = *(const v4f*)(arow + aoff + 4);
        v4f a2 = *(const v4f*)(arow + aoff + 16);
        v4f a3 = *(const v4f*)(arow + aoff + 20);
        v16bf av;
        #pragma unroll
        for (int j = 0; j < 4; ++j) {
            av[j]      = (__bf16)a0[j];
            av[j + 4]  = (__bf16)a1[j];
            av[j + 8]  = (__bf16)a2[j];
            av[j + 12] = (__bf16)a3[j];
        }

        // B 32x16 bf16 layout: lane<16 -> K = {0..7, 16..23}; lane>=16 -> +8
        const int boff = k0 + (hi ? 8 : 0);
        v16bf bv[NT];
        #pragma unroll
        for (int t = 0; t < NT; ++t) {
            v4f b0 = *(const v4f*)(wrow[t] + boff);
            v4f b1 = *(const v4f*)(wrow[t] + boff + 4);
            v4f b2 = *(const v4f*)(wrow[t] + boff + 16);
            v4f b3 = *(const v4f*)(wrow[t] + boff + 20);
            #pragma unroll
            for (int j = 0; j < 4; ++j) {
                bv[t][j]      = (__bf16)b0[j];
                bv[t][j + 4]  = (__bf16)b1[j];
                bv[t][j + 8]  = (__bf16)b2[j];
                bv[t][j + 12] = (__bf16)b3[j];
            }
        }

        #pragma unroll
        for (int t = 0; t < NT; ++t)
            acc[t] = __builtin_amdgcn_wmma_f32_16x16x32_bf16(
                false, av, false, bv[t], (short)0, acc[t], false, false);
    }

    #pragma unroll
    for (int t = 0; t < NT; ++t) {
        float badd = 0.0f;
        if (bias) badd = bias[nBase + t * 16 + l15];
        float* ob = Cout + crow + nBase + t * 16 + l15;
        #pragma unroll
        for (int r = 0; r < 8; ++r) ob[(size_t)r * N] = acc[t][r] + badd;
    }
}

// xi = silu(xz[:, :512] * conv_w[:,0,3] + conv_b)   (L==1: only last conv tap)
__global__ __launch_bounds__(256) void conv_silu_kernel(
    const float* __restrict__ xz, const float* __restrict__ cw,
    const float* __restrict__ cb, float* __restrict__ xi)
{
    const int idx = blockIdx.x * 256 + threadIdx.x;   // B*512 threads
    const int b = idx >> 9, d = idx & 511;
    const float v = xz[(size_t)b * 1024 + d] * cw[d * 4 + 3] + cb[d];
    xi[idx] = v / (1.0f + __expf(-v));
}

// y = xi * (softplus(dt_low @ dtw^T + dtb) * (Bm.Cm) + D) * silu(z)
// (scan with h0=0, L=1:  ys = dt * xi * dot(B, C);  A_log unused)
__global__ __launch_bounds__(256) void ssm_fuse_kernel(
    const float* __restrict__ xz, const float* __restrict__ xi,
    const float* __restrict__ xdbl, const float* __restrict__ dtw,
    const float* __restrict__ dtb, const float* __restrict__ Dp,
    float* __restrict__ y)
{
    const int idx = blockIdx.x * 256 + threadIdx.x;   // B*512 threads
    const int b = idx >> 9, d = idx & 511;
    const float* xr = xdbl + (size_t)b * 48;

    float acc = dtb[d];
    #pragma unroll
    for (int r = 0; r < 16; ++r) acc += xr[r] * dtw[d * 16 + r];
    const float dt = (acc > 20.0f) ? acc : log1pf(__expf(acc));   // softplus

    float bc = 0.0f;
    #pragma unroll
    for (int s = 0; s < 16; ++s) bc += xr[16 + s] * xr[32 + s];   // B . C

    const float zv = xz[(size_t)b * 1024 + 512 + d];
    const float sz = zv / (1.0f + __expf(-zv));                    // silu(z)
    y[idx] = xi[idx] * (dt * bc + Dp[d]) * sz;
}

// LayerNorm(256) + Linear(256 -> 8). One wave per batch row.
__global__ __launch_bounds__(256) void ln_cls_kernel(
    const float* __restrict__ h, const float* __restrict__ g,
    const float* __restrict__ beta, const float* __restrict__ Wc,
    const float* __restrict__ bc, float* __restrict__ out)
{
    const int lane = threadIdx.x & 31;
    const int row  = blockIdx.x * 8 + (threadIdx.x >> 5);
    const float* hr = h + (size_t)row * 256;

    float v[8], sum = 0.0f;
    #pragma unroll
    for (int j = 0; j < 8; ++j) { v[j] = hr[lane + 32 * j]; sum += v[j]; }
    #pragma unroll
    for (int off = 16; off > 0; off >>= 1) sum += __shfl_xor(sum, off, 32);
    const float mu = sum * (1.0f / 256.0f);

    float var = 0.0f;
    #pragma unroll
    for (int j = 0; j < 8; ++j) { const float t = v[j] - mu; var += t * t; }
    #pragma unroll
    for (int off = 16; off > 0; off >>= 1) var += __shfl_xor(var, off, 32);
    const float rs = rsqrtf(var * (1.0f / 256.0f) + 1e-5f);

    float hn[8];
    #pragma unroll
    for (int j = 0; j < 8; ++j)
        hn[j] = (v[j] - mu) * rs * g[lane + 32 * j] + beta[lane + 32 * j];

    #pragma unroll
    for (int c = 0; c < 8; ++c) {
        float p = 0.0f;
        #pragma unroll
        for (int j = 0; j < 8; ++j) p += hn[j] * Wc[c * 256 + lane + 32 * j];
        #pragma unroll
        for (int off = 16; off > 0; off >>= 1) p += __shfl_xor(p, off, 32);
        if (lane == 0) out[(size_t)row * 8 + c] = p + bc[c];
    }
}

extern "C" void kernel_launch(void* const* d_in, const int* in_sizes, int n_in,
                              void* d_out, int out_size, void* d_ws, size_t ws_size,
                              hipStream_t stream) {
    (void)in_sizes; (void)n_in; (void)out_size; (void)ws_size;
    const float* x        = (const float*)d_in[0];   // (8192, 2048)
    const float* W_pre    = (const float*)d_in[1];   // (256, 2048)
    const float* b_pre    = (const float*)d_in[2];   // (256,)
    const float* in_w     = (const float*)d_in[3];   // (5, 1024, 256)
    const float* conv_w   = (const float*)d_in[4];   // (5, 512, 1, 4)
    const float* conv_b   = (const float*)d_in[5];   // (5, 512)
    const float* x_w      = (const float*)d_in[6];   // (5, 48, 512)
    const float* dt_w     = (const float*)d_in[7];   // (5, 512, 16)
    const float* dt_b     = (const float*)d_in[8];   // (5, 512)
    /* d_in[9] = A_log: unused — L==1 scan from h0=0 makes dA irrelevant */
    const float* Dvec     = (const float*)d_in[10];  // (5, 512)
    const float* out_w    = (const float*)d_in[11];  // (5, 256, 512)
    const float* ln_g     = (const float*)d_in[12];  // (256,)
    const float* ln_b     = (const float*)d_in[13];  // (256,)
    const float* W_cls    = (const float*)d_in[14];  // (8, 256)
    const float* b_cls    = (const float*)d_in[15];  // (8,)
    float* out = (float*)d_out;                      // (8192, 8)

    const int B = 8192;
    float* ws   = (float*)d_ws;
    float* h    = ws;                       // B*256
    float* xz   = h    + (size_t)B * 256;   // B*1024
    float* xi   = xz   + (size_t)B * 1024;  // B*512
    float* yb   = xi   + (size_t)B * 512;   // B*512
    float* xdbl = yb   + (size_t)B * 512;   // B*48
    // total: ~77 MB of workspace

    const dim3 blk(256);
    const int mBlocks = B / 128;   // 64

    // h = x @ W_pre^T + b_pre        (N=256, K=2048), 16x64 tile per wave
    gemm_bf16_wmma<4><<<dim3(4, mBlocks), blk, 0, stream>>>(
        x, W_pre, b_pre, nullptr, h, 256, 2048);

    for (int i = 0; i < 5; ++i) {
        const float* inw = in_w   + (size_t)i * 1024 * 256;
        const float* cw  = conv_w + (size_t)i * 512 * 4;
        const float* cb  = conv_b + (size_t)i * 512;
        const float* xw  = x_w    + (size_t)i * 48 * 512;
        const float* dtw = dt_w   + (size_t)i * 512 * 16;
        const float* dtb = dt_b   + (size_t)i * 512;
        const float* Dp  = Dvec   + (size_t)i * 512;
        const float* ow  = out_w  + (size_t)i * 256 * 512;

        // xz = h @ in_w^T              (N=1024, K=256)
        gemm_bf16_wmma<4><<<dim3(16, mBlocks), blk, 0, stream>>>(
            h, inw, nullptr, nullptr, xz, 1024, 256);
        // xi = silu(conv tap)
        conv_silu_kernel<<<B * 512 / 256, blk, 0, stream>>>(xz, cw, cb, xi);
        // xdbl = xi @ x_proj_w^T       (N=48, K=512), NT=3 covers N exactly
        gemm_bf16_wmma<3><<<dim3(1, mBlocks), blk, 0, stream>>>(
            xi, xw, nullptr, nullptr, xdbl, 48, 512);
        // y = xi * (softplus(dt)·(B·C) + D) * silu(z)
        ssm_fuse_kernel<<<B * 512 / 256, blk, 0, stream>>>(
            xz, xi, xdbl, dtw, dtb, Dp, yb);
        // h += y @ out_w^T             (N=256, K=512), residual via C operand
        gemm_bf16_wmma<4><<<dim3(4, mBlocks), blk, 0, stream>>>(
            yb, ow, nullptr, h, h, 256, 512);
    }

    // LayerNorm + classifier head
    ln_cls_kernel<<<B / 8, blk, 0, stream>>>(h, ln_g, ln_b, W_cls, b_cls, out);
}
